// MultiVariantLSTM_68521908240391
// MI455X (gfx1250) — matrix operand order repaired and checked by reading, not compiled, not run on hardware
//
#include <hip/hip_runtime.h>

// MI455X / gfx1250: wave32, WMMA 16x16x32 f16->f32.
typedef __attribute__((ext_vector_type(16))) _Float16 v16h;
typedef __attribute__((ext_vector_type(8)))  _Float16 v8h;
typedef __attribute__((ext_vector_type(8)))  float    v8f;

#define TT 512
#define NB 512
#define FF 64
#define H1 128
#define H2 64

__device__ __forceinline__ float sigmoidf_(float x) { return 1.0f / (1.0f + __expf(-x)); }

__device__ __forceinline__ v8h cvt8_(float4 a, float4 b) {
  v8h r;
  r[0] = (_Float16)a.x; r[1] = (_Float16)a.y; r[2] = (_Float16)a.z; r[3] = (_Float16)a.w;
  r[4] = (_Float16)b.x; r[5] = (_Float16)b.y; r[6] = (_Float16)b.z; r[7] = (_Float16)b.w;
  return r;
}
__device__ __forceinline__ v16h cat16_(v8h lo, v8h hi) {
  return __builtin_shufflevector(lo, hi, 0,1,2,3,4,5,6,7,8,9,10,11,12,13,14,15);
}

// ---------------------------------------------------------------------------
// Pre-swizzle [W;U] (f32, row-major) into WMMA B-fragment layout (f16):
//   frag[nt][kt][lane][e],  n = nt*16 + lane%16,  k = kt*32 + (lane/16)*8 + (e%8) + (e/8)*16
// wu1: [W1(64);U1(128)] x 512 cols -> 32 ntiles * 6 ktiles.  (192 KB)
// wu2: [W2(128);U2(64)] x 256 cols -> 16 ntiles * 6 ktiles.  ( 96 KB)
// ---------------------------------------------------------------------------
__global__ void lstm_swz_weights(const float* __restrict__ W1, const float* __restrict__ U1,
                                 const float* __restrict__ W2, const float* __restrict__ U2,
                                 _Float16* __restrict__ wu1, _Float16* __restrict__ wu2) {
  int idx = blockIdx.x * 256 + threadIdx.x;
  const int N1 = 32 * 6 * 32 * 16;  // 98304
  const int N2 = 16 * 6 * 32 * 16;  // 49152
  if (idx < N1) {
    int e = idx & 15, lane = (idx >> 4) & 31, kt = (idx >> 9) % 6, nt = (idx >> 9) / 6;
    int n = nt * 16 + (lane & 15);
    int k = kt * 32 + (lane >> 4) * 8 + (e & 7) + (e >> 3) * 16;
    float v = (k < FF) ? W1[k * (4 * H1) + n] : U1[(k - FF) * (4 * H1) + n];
    wu1[idx] = (_Float16)v;
  } else if (idx < N1 + N2) {
    int i2 = idx - N1;
    int e = i2 & 15, lane = (i2 >> 4) & 31, kt = (i2 >> 9) % 6, nt = (i2 >> 9) / 6;
    int n = nt * 16 + (lane & 15);
    int k = kt * 32 + (lane >> 4) * 8 + (e & 7) + (e >> 3) * 16;
    float v = (k < H1) ? W2[k * (4 * H2) + n] : U2[(k - H1) * (4 * H2) + n];
    wu2[i2] = (_Float16)v;
  }
}

// ---------------------------------------------------------------------------
// LSTM layer 1: fused [x_t | h_prev] @ [W1;U1] + b1, relu-LSTM update.
// 32 WGs x 16 batch rows; 8 waves each own within-gate cols [16w,16w+16) of
// ALL four gates -> gate math + c state stay in registers across all 512 steps.
// Writes h1 sequence to global as f16 in per-block row-major [bb][t][16][128]
// (exactly the A-fragment source layout layer 2 wants).
// ---------------------------------------------------------------------------
__global__ __launch_bounds__(256) void lstm1_kernel(
    const float* __restrict__ x, const float* __restrict__ b1,
    const _Float16* __restrict__ wu1, _Float16* __restrict__ h1blk) {
  __shared__ _Float16 hbuf[16 * H1];
  const int tid = threadIdx.x;
  const int w = tid >> 5, lane = tid & 31;
  const int bb = blockIdx.x;
  const int m = lane & 15, half = lane >> 4;
  const int j = (w << 4) + m;       // within-gate output column (0..127)
  const int row0 = half * 8;        // C/D fragment row base for this lane

  // Time-invariant B fragments, held in VGPRs: [gate][ktile]
  v16h Bfr[4][6];
#pragma unroll
  for (int g = 0; g < 4; ++g) {
    const _Float16* p = wu1 + (size_t)(g * 8 + w) * 6 * 512 + lane * 16;
#pragma unroll
    for (int kt = 0; kt < 6; ++kt) Bfr[g][kt] = *(const v16h*)(p + kt * 512);
  }
  v8f bias[4];
#pragma unroll
  for (int g = 0; g < 4; ++g) {
    float b = b1[g * H1 + j];
#pragma unroll
    for (int e = 0; e < 8; ++e) bias[g][e] = b;
  }
  for (int k2 = tid; k2 < 16 * H1; k2 += 256) hbuf[k2] = (_Float16)0.f;
  v8f c = {0.f, 0.f, 0.f, 0.f, 0.f, 0.f, 0.f, 0.f};
  __syncthreads();

  const float* xrow = x + (size_t)(bb * 16 + m) * TT * FF;
  for (int t = 0; t < TT; ++t) {
    const float* xt = xrow + (size_t)t * FF;
    if (t + 1 < TT) __builtin_prefetch(xrow + (size_t)(t + 1) * FF + half * 8, 0, 0);
    v8f acc[4] = {bias[0], bias[1], bias[2], bias[3]};
#pragma unroll
    for (int kt = 0; kt < 6; ++kt) {
      v16h a;
      if (kt < 2) {  // K = x features, converted f32->f16 on the fly
        const float* p = xt + kt * 32 + half * 8;
        v8h lo = cvt8_(*(const float4*)p,        *(const float4*)(p + 4));
        v8h hi = cvt8_(*(const float4*)(p + 16), *(const float4*)(p + 20));
        a = cat16_(lo, hi);
      } else {       // K = h_prev from LDS (A-fragment runs are contiguous b128s)
        const _Float16* p = &hbuf[m * H1 + (kt - 2) * 32 + half * 8];
        a = cat16_(*(const v8h*)p, *(const v8h*)(p + 16));
      }
#pragma unroll
      for (int g = 0; g < 4; ++g)
        acc[g] = __builtin_amdgcn_wmma_f32_16x16x32_f16(false, a, false, Bfr[g][kt],
                                                        (short)0, acc[g], false, false);
    }
    __syncthreads();  // all waves done reading h(t-1)
    _Float16* hw = &hbuf[row0 * H1 + j];
#pragma unroll
    for (int e = 0; e < 8; ++e) {
      float iv = sigmoidf_(acc[0][e]);
      float fv = sigmoidf_(acc[1][e]);
      float gv = fmaxf(acc[2][e], 0.0f);   // candidate: relu
      float ov = sigmoidf_(acc[3][e]);
      float cv = fv * c[e] + iv * gv;
      c[e] = cv;
      float hv = ov * fmaxf(cv, 0.0f);     // output: relu
      hw[e * H1] = (_Float16)hv;
    }
    __syncthreads();  // h(t) complete in LDS
    // cooperative coalesced LDS -> global copy of h(t) (16B per thread)
    *(v8h*)(h1blk + ((size_t)bb * TT + t) * (16 * H1) + tid * 8) =
        *(const v8h*)(hbuf + tid * 8);
  }
}

// ---------------------------------------------------------------------------
// LSTM layer 2: fused [h1_t | h_prev] @ [W2;U2] + b2; only final hT kept.
// 32 WGs x 4 waves (gate width 64 -> 4 ntiles each).
// ---------------------------------------------------------------------------
__global__ __launch_bounds__(128) void lstm2_kernel(
    const _Float16* __restrict__ h1blk, const float* __restrict__ b2,
    const _Float16* __restrict__ wu2, float* __restrict__ hT) {
  __shared__ _Float16 hbuf[16 * H2];
  const int tid = threadIdx.x;
  const int w = tid >> 5, lane = tid & 31;
  const int bb = blockIdx.x;
  const int m = lane & 15, half = lane >> 4;
  const int j = (w << 4) + m;
  const int row0 = half * 8;

  v16h Bfr[4][6];
#pragma unroll
  for (int g = 0; g < 4; ++g) {
    const _Float16* p = wu2 + (size_t)(g * 4 + w) * 6 * 512 + lane * 16;
#pragma unroll
    for (int kt = 0; kt < 6; ++kt) Bfr[g][kt] = *(const v16h*)(p + kt * 512);
  }
  v8f bias[4];
#pragma unroll
  for (int g = 0; g < 4; ++g) {
    float b = b2[g * H2 + j];
#pragma unroll
    for (int e = 0; e < 8; ++e) bias[g][e] = b;
  }
  for (int k2 = tid; k2 < 16 * H2; k2 += 128) hbuf[k2] = (_Float16)0.f;
  v8f c = {0.f, 0.f, 0.f, 0.f, 0.f, 0.f, 0.f, 0.f};
  __syncthreads();

  const _Float16* h1b = h1blk + (size_t)bb * TT * (16 * H1);
  for (int t = 0; t < TT; ++t) {
    const _Float16* h1t = h1b + (size_t)t * (16 * H1);
    if (t + 1 < TT) __builtin_prefetch(h1t + (16 * H1) + m * H1 + half * 8, 0, 0);
    v8f acc[4] = {bias[0], bias[1], bias[2], bias[3]};
#pragma unroll
    for (int kt = 0; kt < 6; ++kt) {
      v16h a;
      if (kt < 4) {  // K = layer-1 hidden state (f16 block layout from lstm1)
        const _Float16* p = h1t + m * H1 + kt * 32 + half * 8;
        a = cat16_(*(const v8h*)p, *(const v8h*)(p + 16));
      } else {       // K = h_prev from LDS
        const _Float16* p = &hbuf[m * H2 + (kt - 4) * 32 + half * 8];
        a = cat16_(*(const v8h*)p, *(const v8h*)(p + 16));
      }
#pragma unroll
      for (int g = 0; g < 4; ++g)
        acc[g] = __builtin_amdgcn_wmma_f32_16x16x32_f16(false, a, false, Bfr[g][kt],
                                                        (short)0, acc[g], false, false);
    }
    __syncthreads();
    _Float16* hw = &hbuf[row0 * H2 + j];
#pragma unroll
    for (int e = 0; e < 8; ++e) {
      float iv = sigmoidf_(acc[0][e]);
      float fv = sigmoidf_(acc[1][e]);
      float gv = fmaxf(acc[2][e], 0.0f);
      float ov = sigmoidf_(acc[3][e]);
      float cv = fv * c[e] + iv * gv;
      c[e] = cv;
      float hv = ov * fmaxf(cv, 0.0f);
      hw[e * H2] = (_Float16)hv;
      if (t == TT - 1) hT[(size_t)(bb * 16 + row0 + e) * H2 + j] = hv;
    }
    __syncthreads();
  }
}

// ---------------------------------------------------------------------------
// Dense head: [512,64] @ [64,25] + b -> @ [25,1] + b.  ~1.6 MFLOP: plain VALU.
// ---------------------------------------------------------------------------
__global__ void lstm_dense_kernel(const float* __restrict__ hT,
                                  const float* __restrict__ Wd1, const float* __restrict__ bd1,
                                  const float* __restrict__ Wd2, const float* __restrict__ bd2,
                                  float* __restrict__ out) {
  int b = blockIdx.x * 256 + threadIdx.x;
  if (b >= NB) return;
  const float* h = hT + (size_t)b * H2;
  float acc = bd2[0];
  for (int d = 0; d < 25; ++d) {
    float s = bd1[d];
    for (int k = 0; k < H2; ++k) s += h[k] * Wd1[k * 25 + d];
    acc += s * Wd2[d];
  }
  out[b] = acc;
}

extern "C" void kernel_launch(void* const* d_in, const int* in_sizes, int n_in,
                              void* d_out, int out_size, void* d_ws, size_t ws_size,
                              hipStream_t stream) {
  const float* x   = (const float*)d_in[0];
  const float* W1  = (const float*)d_in[1];
  const float* U1  = (const float*)d_in[2];
  const float* b1  = (const float*)d_in[3];
  const float* W2  = (const float*)d_in[4];
  const float* U2  = (const float*)d_in[5];
  const float* b2  = (const float*)d_in[6];
  const float* Wd1 = (const float*)d_in[7];
  const float* bd1 = (const float*)d_in[8];
  const float* Wd2 = (const float*)d_in[9];
  const float* bd2 = (const float*)d_in[10];

  // Workspace layout (bytes):
  //   [0,196608)            wu1 swizzled f16 [W1;U1]
  //   [196608,294912)       wu2 swizzled f16 [W2;U2]
  //   [294912,+64MB)        h1 sequence, f16 block layout [bb][t][16][128]
  //   then                  hT f32 [512][64]
  char* ws = (char*)d_ws;
  _Float16* wu1   = (_Float16*)ws;
  _Float16* wu2   = (_Float16*)(ws + 196608);
  _Float16* h1blk = (_Float16*)(ws + 294912);
  float*    hT    = (float*)(ws + 294912 + (size_t)67108864);
  float*    out   = (float*)d_out;

  lstm_swz_weights<<<(98304 + 49152 + 255) / 256, 256, 0, stream>>>(W1, U1, W2, U2, wu1, wu2);
  lstm1_kernel<<<NB / 16, 256, 0, stream>>>(x, b1, wu1, h1blk);
  lstm2_kernel<<<NB / 16, 128, 0, stream>>>(h1blk, b2, wu2, hT);
  lstm_dense_kernel<<<(NB + 255) / 256, 256, 0, stream>>>(hT, Wd1, bd1, Wd2, bd2, out);
}
